// Attention_15058155340425
// MI455X (gfx1250) — compile-verified, matrix operand order checked by reading, not run on hardware
//
#include <hip/hip_runtime.h>
#include <hip/hip_bf16.h>

typedef __bf16 bf16;
typedef __bf16 v16bf __attribute__((ext_vector_type(16)));
typedef __bf16 v8bf  __attribute__((ext_vector_type(8)));
typedef float  v8f   __attribute__((ext_vector_type(8)));
typedef int    v4i   __attribute__((ext_vector_type(4)));

#define DIM     768
#define NHEADS  12
#define HDIM    64
#define SEQ     1024
#define BATCH   8
#define MTOT    (BATCH * SEQ)     /* 8192 tokens */
#define NQKV    (3 * DIM)         /* 2304 */
#define QK_SCALE 0.125f           /* 64^-0.5 */

#define AS1 __attribute__((address_space(1)))
#define AS3 __attribute__((address_space(3)))

#if defined(__AMDGCN__) && __has_builtin(__builtin_amdgcn_global_load_async_to_lds_b128) && \
    __has_builtin(__builtin_amdgcn_s_wait_asynccnt)
#define HAVE_ASYNC_LDS 1
#else
#define HAVE_ASYNC_LDS 0
#endif

static __device__ __forceinline__ v16bf ld16(const bf16* p) {
    return *(const v16bf*)p;
}

static __device__ __forceinline__ v8f wmma_bf16(v16bf a, v16bf b, v8f c) {
    // D = A(16x32 bf16) x B(32x16 bf16) + C(16x16 f32)
    return __builtin_amdgcn_wmma_f32_16x16x32_bf16(
        false, a, false, b, (short)0, c, false, false);
}

// 16-byte global -> LDS copy; async (ASYNCcnt, no VGPR round trip).
static __device__ __forceinline__ void cp16_g2l(const bf16* g, bf16* l) {
#if HAVE_ASYNC_LDS
    __builtin_amdgcn_global_load_async_to_lds_b128(
        (AS1 v4i*)(unsigned long long)g,
        (AS3 v4i*)(unsigned)(unsigned long long)l, 0, 0);
#else
    *(v8bf*)l = *(const v8bf*)g;
#endif
}
static __device__ __forceinline__ void cp_wait() {
#if HAVE_ASYNC_LDS
    __builtin_amdgcn_s_wait_asynccnt(0);
#endif
}

// ---------------------------------------------------------------- converts
__global__ void cvt_cast_kernel(const float* __restrict__ in,
                                bf16* __restrict__ out, int n) {
    int i = blockIdx.x * blockDim.x + threadIdx.x;
    if (i < n) out[i] = (bf16)in[i];
}

// out[c][r] = in[r][c]  (K-major weights for the WMMA B-operand)
__global__ void cvt_transpose_kernel(const float* __restrict__ in,
                                     bf16* __restrict__ out,
                                     int rows, int cols) {
    int i = blockIdx.x * blockDim.x + threadIdx.x;
    if (i < rows * cols) {
        int r = i / cols, c = i % cols;
        out[c * rows + r] = (bf16)in[i];
    }
}

// ---------------------------------------------------------------- QKV GEMM
// 64x64 wave tile, ping-pong unroll-by-2 over K: two disjoint fragment
// register sets, no rotation copies; 32 wmma per loop body.
__global__ __launch_bounds__(256) void qkv_gemm_kernel(
    const bf16* __restrict__ xb, const bf16* __restrict__ wt,
    bf16* __restrict__ qws, bf16* __restrict__ kws, bf16* __restrict__ vtws) {
    const int wid   = threadIdx.x >> 5;
    const int lane  = threadIdx.x & 31;
    const int lrow  = lane & 15;
    const int lhalf = lane >> 4;
    const int wave  = blockIdx.x * 8 + wid;
    const int nt    = wave % (NQKV / 64);   // 0..35
    const int mt    = wave / (NQKV / 64);   // 0..127

    const bf16* abase = xb + (mt * 64 + lrow) * DIM + lhalf * 16;
    const bf16* bbase = wt + (nt * 64 + lrow) * DIM + lhalf * 16;

    v16bf a0[4], b0[4];
#pragma unroll
    for (int r = 0; r < 4; ++r) a0[r] = ld16(abase + r * 16 * DIM);
#pragma unroll
    for (int s = 0; s < 4; ++s) b0[s] = ld16(bbase + s * 16 * DIM);

    v8f acc[4][4] = {};
#pragma unroll 1
    for (int k0 = 0; k0 < DIM; k0 += 64) {
        v16bf a1[4], b1[4];
        // prefetch odd half (k0+32 < DIM always holds: DIM % 64 == 0)
#pragma unroll
        for (int r = 0; r < 4; ++r) a1[r] = ld16(abase + r * 16 * DIM + k0 + 32);
#pragma unroll
        for (int s = 0; s < 4; ++s) b1[s] = ld16(bbase + s * 16 * DIM + k0 + 32);
#pragma unroll
        for (int r = 0; r < 4; ++r)
#pragma unroll
            for (int s = 0; s < 4; ++s)
                acc[r][s] = wmma_bf16(a0[r], b0[s], acc[r][s]);
        if (k0 + 64 < DIM) {   // prefetch next even half
#pragma unroll
            for (int r = 0; r < 4; ++r) a0[r] = ld16(abase + r * 16 * DIM + k0 + 64);
#pragma unroll
            for (int s = 0; s < 4; ++s) b0[s] = ld16(bbase + s * 16 * DIM + k0 + 64);
        }
#pragma unroll
        for (int r = 0; r < 4; ++r)
#pragma unroll
            for (int s = 0; s < 4; ++s)
                acc[r][s] = wmma_bf16(a1[r], b1[s], acc[r][s]);
    }

    // epilogue: 64 tile columns lie entirely inside one (which, head) section
    const int which = nt / NHEADS;          // 0=Q 1=K 2=V
    const int h     = nt % NHEADS;
    const int b0i   = (mt * 64) >> 10;      // batch, constant across the tile
    const int bh    = b0i * NHEADS + h;
#pragma unroll
    for (int r = 0; r < 4; ++r) {
        const int ns0 = (mt * 64 + r * 16 + 8 * lhalf) & (SEQ - 1);
        if (which == 0) {                   // Q, pre-scaled
#pragma unroll
            for (int s = 0; s < 4; ++s) {
                int d   = s * 16 + lrow;
                bf16* p = qws + (bh * SEQ + ns0) * HDIM + d;
#pragma unroll
                for (int i = 0; i < 8; ++i) p[i * HDIM] = (bf16)(acc[r][s][i] * QK_SCALE);
            }
        } else if (which == 1) {            // K row-major [n][d]
#pragma unroll
            for (int s = 0; s < 4; ++s) {
                int d   = s * 16 + lrow;
                bf16* p = kws + (bh * SEQ + ns0) * HDIM + d;
#pragma unroll
                for (int i = 0; i < 8; ++i) p[i * HDIM] = (bf16)acc[r][s][i];
            }
        } else {                            // V transposed [d][n]: packed b128 store
#pragma unroll
            for (int s = 0; s < 4; ++s) {
                int d = s * 16 + lrow;
                v8bf pk;
#pragma unroll
                for (int i = 0; i < 8; ++i) pk[i] = (bf16)acc[r][s][i];
                *(v8bf*)(vtws + (bh * HDIM + d) * SEQ + ns0) = pk;
            }
        }
    }
}

// ---------------------------------------------------------------- attention
// 8 waves share (b,h): K/V chunks staged once per block into LDS via
// async-to-LDS DMA, double-buffered against compute. One wave = 16 query rows.
__global__ __launch_bounds__(256) void attn_kernel(
    const bf16* __restrict__ qws, const bf16* __restrict__ kws,
    const bf16* __restrict__ vtws, bf16* __restrict__ ows) {
    __shared__ __attribute__((aligned(64))) bf16 ldsK[2][32 * HDIM]; // [key][d]
    __shared__ __attribute__((aligned(64))) bf16 ldsV[2][HDIM * 32]; // [d][key]
    __shared__ __attribute__((aligned(64))) bf16 plds[8][16 * 32];

    const int tid   = threadIdx.x;
    const int wid   = tid >> 5;
    const int lane  = tid & 31;
    const int lrow  = lane & 15;
    const int lhalf = lane >> 4;
    const int bh    = blockIdx.x >> 3;
    const int rb    = blockIdx.x & 7;
    const int qbase = rb * 128 + wid * 16;
    bf16* myp = plds[wid];

    // staging source pointers (whole block cooperates; 16B per thread per tile)
    const bf16* ksrc = kws + bh * SEQ * HDIM;          // chunk kc at + kc*HDIM
    const bf16* vsrc = vtws + bh * HDIM * SEQ;
    const int vrow = tid >> 2;                         // 0..63
    const int vcol = (tid & 3) * 8;                    // 0,8,16,24

    // Q fragments, reused for every key chunk
    const bf16* qrow = qws + (bh * SEQ + qbase + lrow) * HDIM;
    v16bf va0 = ld16(qrow + lhalf * 16);
    v16bf va1 = ld16(qrow + 32 + lhalf * 16);

    float mstate[8], lstate[8];
#pragma unroll
    for (int i = 0; i < 8; ++i) { mstate[i] = -3.0e38f; lstate[i] = 0.f; }
    v8f o[4] = {};

    // prologue: stage chunk 0 into buffer 0
    cp16_g2l(ksrc + tid * 8, &ldsK[0][tid * 8]);
    cp16_g2l(vsrc + vrow * SEQ + vcol, &ldsV[0][vrow * 32 + vcol]);
    cp_wait();
    __syncthreads();

    for (int kc = 0; kc < SEQ; kc += 32) {
        const int buf = (kc >> 5) & 1;
        if (kc + 32 < SEQ) {   // kick off next chunk's DMA into the other buffer
            cp16_g2l(ksrc + (kc + 32) * HDIM + tid * 8, &ldsK[buf ^ 1][tid * 8]);
            cp16_g2l(vsrc + vrow * SEQ + (kc + 32) + vcol, &ldsV[buf ^ 1][vrow * 32 + vcol]);
        }

        // S = Q*Kt for 32 keys -> two 16x16 f32 tiles (K frags from LDS)
        const bf16* kb = ldsK[buf];
        v8f s0 = {}, s1 = {};
        s0 = wmma_bf16(va0, ld16(kb + lrow * HDIM + lhalf * 16), s0);
        s0 = wmma_bf16(va1, ld16(kb + lrow * HDIM + 32 + lhalf * 16), s0);
        s1 = wmma_bf16(va0, ld16(kb + (16 + lrow) * HDIM + lhalf * 16), s1);
        s1 = wmma_bf16(va1, ld16(kb + (16 + lrow) * HDIM + 32 + lhalf * 16), s1);

        float alpha[8];
#pragma unroll
        for (int i = 0; i < 8; ++i) {
            float v = fmaxf(s0[i], s1[i]);
            v = fmaxf(v, __shfl_xor(v, 1));
            v = fmaxf(v, __shfl_xor(v, 2));
            v = fmaxf(v, __shfl_xor(v, 4));
            v = fmaxf(v, __shfl_xor(v, 8));
            float nm = fmaxf(mstate[i], v);
            float p0 = __expf(s0[i] - nm);
            float p1 = __expf(s1[i] - nm);
            myp[(i + 8 * lhalf) * 32 + lrow]      = (bf16)p0;
            myp[(i + 8 * lhalf) * 32 + 16 + lrow] = (bf16)p1;
            float rs = p0 + p1;
            rs += __shfl_xor(rs, 1);
            rs += __shfl_xor(rs, 2);
            rs += __shfl_xor(rs, 4);
            rs += __shfl_xor(rs, 8);
            float a = __expf(mstate[i] - nm);
            lstate[i] = lstate[i] * a + rs;
            mstate[i] = nm;
            alpha[i]  = a;
        }
#pragma unroll
        for (int t = 0; t < 4; ++t)
#pragma unroll
            for (int i = 0; i < 8; ++i) o[t][i] *= alpha[i];

        __builtin_amdgcn_wave_barrier();
        v16bf pa = ld16(myp + lrow * 32 + lhalf * 16);   // P as A fragment
        __builtin_amdgcn_wave_barrier();

        // O += P * V (V frags from LDS, keys contiguous)
        const bf16* vb = ldsV[buf];
#pragma unroll
        for (int t = 0; t < 4; ++t)
            o[t] = wmma_bf16(pa, ld16(vb + (t * 16 + lrow) * 32 + lhalf * 16), o[t]);

        cp_wait();          // next chunk's DMA landed
        __syncthreads();    // all waves done reading buf; buf reusable
    }

    // normalize + store to [b][n][h*64+d] (row-major A operand for proj GEMM)
    float rl[8];
#pragma unroll
    for (int i = 0; i < 8; ++i) rl[i] = 1.f / lstate[i];
    const int b = bh / NHEADS, h = bh % NHEADS;
#pragma unroll
    for (int t = 0; t < 4; ++t) {
        int d   = h * HDIM + t * 16 + lrow;
        bf16* p = ows + (b * SEQ + qbase + 8 * lhalf) * DIM + d;
#pragma unroll
        for (int i = 0; i < 8; ++i) p[i * DIM] = (bf16)(o[t][i] * rl[i]);
    }
}

// ---------------------------------------------------------------- proj GEMM
// 32x64 wave tile, ping-pong unroll-by-2 over K.
__global__ __launch_bounds__(256) void proj_gemm_kernel(
    const bf16* __restrict__ ob, const bf16* __restrict__ wt,
    const float* __restrict__ bias, float* __restrict__ out) {
    const int wid   = threadIdx.x >> 5;
    const int lane  = threadIdx.x & 31;
    const int lrow  = lane & 15;
    const int lhalf = lane >> 4;
    const int wave  = blockIdx.x * 8 + wid;
    const int nt    = wave % (DIM / 64);   // 0..11
    const int mt    = wave / (DIM / 64);   // 0..255

    const bf16* abase = ob + (mt * 32 + lrow) * DIM + lhalf * 16;
    const bf16* bbase = wt + (nt * 64 + lrow) * DIM + lhalf * 16;

    v16bf a0[2], b0[4];
#pragma unroll
    for (int r = 0; r < 2; ++r) a0[r] = ld16(abase + r * 16 * DIM);
#pragma unroll
    for (int s = 0; s < 4; ++s) b0[s] = ld16(bbase + s * 16 * DIM);

    v8f acc[2][4] = {};
#pragma unroll 1
    for (int k0 = 0; k0 < DIM; k0 += 64) {
        v16bf a1[2], b1[4];
#pragma unroll
        for (int r = 0; r < 2; ++r) a1[r] = ld16(abase + r * 16 * DIM + k0 + 32);
#pragma unroll
        for (int s = 0; s < 4; ++s) b1[s] = ld16(bbase + s * 16 * DIM + k0 + 32);
#pragma unroll
        for (int r = 0; r < 2; ++r)
#pragma unroll
            for (int s = 0; s < 4; ++s)
                acc[r][s] = wmma_bf16(a0[r], b0[s], acc[r][s]);
        if (k0 + 64 < DIM) {
#pragma unroll
            for (int r = 0; r < 2; ++r) a0[r] = ld16(abase + r * 16 * DIM + k0 + 64);
#pragma unroll
            for (int s = 0; s < 4; ++s) b0[s] = ld16(bbase + s * 16 * DIM + k0 + 64);
        }
#pragma unroll
        for (int r = 0; r < 2; ++r)
#pragma unroll
            for (int s = 0; s < 4; ++s)
                acc[r][s] = wmma_bf16(a1[r], b1[s], acc[r][s]);
    }

#pragma unroll
    for (int r = 0; r < 2; ++r) {
        const int rowbase = mt * 32 + r * 16 + 8 * lhalf;
#pragma unroll
        for (int s = 0; s < 4; ++s) {
            int c    = nt * 64 + s * 16 + lrow;
            float bv = bias[c];
            float* p = out + rowbase * DIM + c;
#pragma unroll
            for (int i = 0; i < 8; ++i) p[i * DIM] = acc[r][s][i] + bv;
        }
    }
}

// ---------------------------------------------------------------- launcher
extern "C" void kernel_launch(void* const* d_in, const int* in_sizes, int n_in,
                              void* d_out, int out_size, void* d_ws, size_t ws_size,
                              hipStream_t stream) {
    const float* x      = (const float*)d_in[0];
    const float* w_qkv  = (const float*)d_in[1];
    const float* w_proj = (const float*)d_in[2];
    const float* b_proj = (const float*)d_in[3];
    float* out = (float*)d_out;

    char* ws = (char*)d_ws;
    size_t off = 0;
    bf16* xb     = (bf16*)(ws + off); off += (size_t)MTOT * DIM * 2;   // 12.6 MB
    bf16* wqkvT  = (bf16*)(ws + off); off += (size_t)NQKV * DIM * 2;   //  3.5 MB
    bf16* wprojT = (bf16*)(ws + off); off += (size_t)DIM * DIM * 2;    //  1.2 MB
    bf16* qws    = (bf16*)(ws + off); off += (size_t)MTOT * DIM * 2;   // 12.6 MB
    bf16* kws    = (bf16*)(ws + off); off += (size_t)MTOT * DIM * 2;   // 12.6 MB
    bf16* vtws   = (bf16*)(ws + off); off += (size_t)MTOT * DIM * 2;   // 12.6 MB
    bf16* ob     = (bf16*)(ws + off); off += (size_t)MTOT * DIM * 2;   // 12.6 MB

    int nx = MTOT * DIM;
    cvt_cast_kernel<<<(nx + 255) / 256, 256, 0, stream>>>(x, xb, nx);
    cvt_transpose_kernel<<<(DIM * NQKV + 255) / 256, 256, 0, stream>>>(w_qkv, wqkvT, DIM, NQKV);
    cvt_transpose_kernel<<<(DIM * DIM + 255) / 256, 256, 0, stream>>>(w_proj, wprojT, DIM, DIM);

    // 128 m-tiles(64) * 36 n-tiles(64), 8 waves/block
    qkv_gemm_kernel<<<(128 * 36) / 8, 256, 0, stream>>>(xb, wqkvT, qws, kws, vtws);
    // 96 (b,h) * 8 row-blocks of 128 query rows
    attn_kernel<<<BATCH * NHEADS * 8, 256, 0, stream>>>(qws, kws, vtws, ob);
    // 256 m-tiles(32) * 12 n-tiles(64), 8 waves/block
    proj_gemm_kernel<<<(256 * 12) / 8, 256, 0, stream>>>(ob, wprojT, b_proj, out);
}
